// Attention_67199058313374
// MI455X (gfx1250) — matrix operand order repaired and checked by reading, not compiled, not run on hardware
//
#include <hip/hip_runtime.h>
#include <hip/hip_bf16.h>
#include <stdint.h>

typedef _Float16 half_t;
typedef __attribute__((ext_vector_type(16))) _Float16 v16h;
typedef __attribute__((ext_vector_type(8)))  _Float16 v8h;
typedef __attribute__((ext_vector_type(8)))  float    v8f;
typedef __attribute__((ext_vector_type(4)))  unsigned int u32x4;
typedef __attribute__((ext_vector_type(8)))  int          i32x8;
typedef __attribute__((ext_vector_type(4)))  int          i32x4;

__device__ __forceinline__ v16h cat8(v8h lo, v8h hi) {
    return __builtin_shufflevector(lo, hi, 0,1,2,3,4,5,6,7,8,9,10,11,12,13,14,15);
}

__device__ __forceinline__ v8f wmma_f16(v16h a, v16h b, v8f c) {
    // D = A*B + C,  v_wmma_f32_16x16x32_f16
    return __builtin_amdgcn_wmma_f32_16x16x32_f16(false, a, false, b, (short)0, c, false, false);
}

// ---------------------------------------------------------------------------
// TDM: 2D f16 tile, global (row-major, elem stride0) -> LDS with row padding.
//   rows x cols halfs; LDS pitch = cols + pad, pad encoded per D# (§8.4).
// Issue from ONE wave; caller waits s_wait_tensorcnt then barriers.
// ---------------------------------------------------------------------------
__device__ __forceinline__ void tdm_load_2d_f16(
    const half_t* gsrc, unsigned lds_byte_addr,
    unsigned tdim0, unsigned tdim1, unsigned stride0,
    unsigned tile0, unsigned tile1,
    unsigned pad_interval_code, unsigned pad_amount_code)
{
    const unsigned long long ga = (unsigned long long)(uintptr_t)gsrc;
    u32x4 g0;
    g0[0] = 1u;                                   // count=1, user descriptor
    g0[1] = lds_byte_addr;                        // lds_addr
    g0[2] = (unsigned)(ga & 0xffffffffu);         // global_addr[31:0]
    g0[3] = (unsigned)((ga >> 32) & 0x01ffffffu)  // global_addr[56:32]
          | (2u << 30);                           // type = 2 ("image")
    i32x8 g1;
    g1[0] = (int)((1u << 16)                      // data_size = 2B
          | (1u << 20)                            // pad_enable
          | (pad_interval_code << 22)
          | (pad_amount_code << 25));
    g1[1] = (int)(tdim0 << 16);                   // tensor_dim0[15:0]
    g1[2] = (int)((tdim0 >> 16) | (tdim1 << 16)); // tensor_dim0 hi | tensor_dim1 lo
    g1[3] = (int)((tdim1 >> 16) | (tile0 << 16)); // tensor_dim1 hi | tile_dim0
    g1[4] = (int)tile1;                           // tile_dim1 (tile_dim2 = 0)
    g1[5] = (int)stride0;                         // tensor_dim0_stride[31:0]
    g1[6] = 0;                                    // stride0 hi | dim1_stride lo
    g1[7] = 0;                                    // dim1_stride hi
    i32x4 z4 = {0, 0, 0, 0};
#if defined(__clang_major__) && __clang_major__ >= 23
    i32x8 z8 = {0, 0, 0, 0, 0, 0, 0, 0};
    __builtin_amdgcn_tensor_load_to_lds(g0, g1, z4, z4, z8, 0);
#else
    __builtin_amdgcn_tensor_load_to_lds(g0, g1, z4, z4, 0);
#endif
}

// ---------------------------------------------------------------------------
// Kernel 1: fused QKV projection.
//   Y = x(131072x256) @ [Wq | Wkv](256x768) + [bq | bkv]
// Block: 256 threads (8 waves, 4Mw x 2Nw), tile 64 rows x 64 cols, K=256.
// Each wave computes 16x32 (2 N-tiles sharing the A fragment).
// ---------------------------------------------------------------------------
#define LDK 264   // padded half-stride for 256-wide K panels (bank-conflict free)

__global__ __launch_bounds__(256) void qkv_kernel(
    const float* __restrict__ x,
    const float* __restrict__ Wq,  const float* __restrict__ bq,
    const float* __restrict__ Wkv, const float* __restrict__ bkv,
    half_t* __restrict__ Q, half_t* __restrict__ K, half_t* __restrict__ V)
{
    __shared__ half_t xs [64 * LDK];   // A tile, row-major (rows x K)
    __shared__ half_t wsT[64 * LDK];   // B tile, transposed (cols x K)

    const int tid  = threadIdx.x;
    const int lane = tid & 31;
    const int w    = tid >> 5;
    const int g    = lane >> 4;
    const int l16  = lane & 15;
    const long rowBase = (long)blockIdx.x * 64;
    const int  colBase = blockIdx.y * 64;

    // stage x tile (f32 -> f16), 64x256, float4-vectorized
    #pragma unroll
    for (int i = 0; i < 16; ++i) {
        int idx = tid + i * 256;           // 4096 float4 chunks
        int r   = idx >> 6;
        int jc  = idx & 63;
        const float4 f = *(const float4*)(x + (rowBase + r) * 256 + jc * 4);
        half_t* p = &xs[r * LDK + jc * 4];
        p[0] = (half_t)f.x; p[1] = (half_t)f.y; p[2] = (half_t)f.z; p[3] = (half_t)f.w;
    }
    // stage W^T tile: 64 cols x 256 K (f32 -> f16, transpose)
    const float* Wsrc; int ldw, coff;
    if (colBase < 256) { Wsrc = Wq;  ldw = 256; coff = colBase; }
    else               { Wsrc = Wkv; ldw = 512; coff = colBase - 256; }
    #pragma unroll
    for (int i = 0; i < 64; ++i) {
        int idx = tid + i * 256;           // 16384 elements
        int k   = idx >> 6;
        int col = idx & 63;
        wsT[col * LDK + k] = (half_t)Wsrc[(long)k * ldw + coff + col];
    }
    __syncthreads();

    const int wm = w >> 1, wn = w & 1;
    const int mBase = wm * 16, nBase = wn * 32;
    const int gcol0 = colBase + nBase + l16;

    float bias0 = (gcol0 < 256) ? bq[gcol0] : bkv[gcol0 - 256];
    float bias1 = (gcol0 + 16 < 256) ? bq[gcol0 + 16] : bkv[gcol0 + 16 - 256];
    v8f acc0 = { bias0,bias0,bias0,bias0,bias0,bias0,bias0,bias0 };
    v8f acc1 = { bias1,bias1,bias1,bias1,bias1,bias1,bias1,bias1 };

    const half_t* arow  = &xs[(mBase + l16) * LDK];
    const half_t* brow0 = &wsT[(nBase + l16) * LDK];
    const half_t* brow1 = &wsT[(nBase + 16 + l16) * LDK];
    #pragma unroll
    for (int kk = 0; kk < 8; ++kk) {
        const int k0 = kk * 32;
        v16h a  = cat8(*(const v8h*)(arow + k0 + g * 8),
                       *(const v8h*)(arow + k0 + 16 + g * 8));
        v16h b0 = cat8(*(const v8h*)(brow0 + k0 + g * 8),
                       *(const v8h*)(brow0 + k0 + 16 + g * 8));
        v16h b1 = cat8(*(const v8h*)(brow1 + k0 + g * 8),
                       *(const v8h*)(brow1 + k0 + 16 + g * 8));
        acc0 = wmma_f16(a, b0, acc0);
        acc1 = wmma_f16(a, b1, acc1);
    }

    const float scale = 0.17677669529663687f;   // 1/sqrt(d=32)
    #pragma unroll
    for (int t = 0; t < 2; ++t) {
        const int gcol = gcol0 + t * 16;
        const v8f acc = t ? acc1 : acc0;
        const bool isQ = gcol < 256;
        const int  c2  = isQ ? 0 : (gcol - 256);
        const bool isK = (!isQ) && (c2 < 256);
        const int  cc  = isQ ? gcol : (c2 & 255);
        const int  head = cc >> 5, dd = cc & 31;
        half_t* dst = isQ ? Q : (isK ? K : V);
        #pragma unroll
        for (int i = 0; i < 8; ++i) {
            long gr = rowBase + mBase + g * 8 + i;   // C layout: M = i + g*8
            long bI = gr >> 8;
            int  n  = (int)(gr & 255);
            float val = acc[i];
            if (isQ) val *= scale;
            dst[((bI * 8 + head) * 256 + n) * 32 + dd] = (half_t)val;
        }
    }
}

// ---------------------------------------------------------------------------
// Kernel 2: fused masked attention per (b,h), flash-style online softmax.
//   S = Qs.K^T + mask (mask rides in the WMMA C operand), O = softmax(S).V
// K tile staged into padded LDS by the Tensor Data Mover (wave 0 issues,
// waits TENSORcnt, workgroup barrier releases).  V staged transposed by VALU.
// ---------------------------------------------------------------------------
#define LDKS 40    // K tile half-stride (32 + 8 pad = 64B + 16B)
#define LDVT 264   // V^T half-stride   (256 + 8 pad)
#define LDPS 40    // P scratch half-stride

__global__ __launch_bounds__(256) void attn_kernel(
    const half_t* __restrict__ Q, const half_t* __restrict__ Kt,
    const half_t* __restrict__ Vt, const float* __restrict__ mask,
    half_t* __restrict__ O)
{
    __shared__ half_t Ks [256 * LDKS];   // (key, d) row-major, TDM-filled
    __shared__ half_t VTs[ 32 * LDVT];   // (d, key) transposed
    __shared__ half_t Ps [8 * 16 * LDPS]; // per-wave P round-trip scratch

    const int tid  = threadIdx.x;
    const int lane = tid & 31;
    const int w    = tid >> 5;
    const int g    = lane >> 4;
    const int l16  = lane & 15;
    const int bh   = blockIdx.x;
    const long base = (long)bh * 256 * 32;
    const int b = bh >> 3, h = bh & 7;
    const float* mbase = mask + (long)(b & 255) * 65536;

    // TDM: K tile 256 keys x 32 halfs, 64B rows padded to 80B LDS pitch.
    if (w == 0) {
        tdm_load_2d_f16(Kt + base,
                        (unsigned)(uintptr_t)(void*)&Ks[0],
                        /*tdim0=*/32, /*tdim1=*/256, /*stride0=*/32,
                        /*tile0=*/32, /*tile1=*/256,
                        /*pad_interval(16 dw)=*/3u, /*pad_amount(4 dw)=*/3u);
        __builtin_amdgcn_s_wait_tensorcnt(0);
    }
    // stage V transposed (one key row per thread)
    {
        const int key = tid;
        const half_t* vr = Vt + base + (long)key * 32;
        #pragma unroll
        for (int d = 0; d < 32; ++d)
            VTs[d * LDVT + key] = vr[d];
    }
    __syncthreads();

    half_t* myPs = &Ps[w * 16 * LDPS];

    for (int qb = 0; qb < 2; ++qb) {
        const int q0 = qb * 128 + w * 16;
        // Q A-fragment: rows q0..q0+15, K = d = 0..31 (loop invariant)
        const half_t* qrow = Q + base + (long)(q0 + l16) * 32;
        const v16h afrag = cat8(*(const v8h*)(qrow + g * 8),
                                *(const v8h*)(qrow + 16 + g * 8));

        v8f o0 = {0,0,0,0,0,0,0,0}, o1 = {0,0,0,0,0,0,0,0};
        float mrow[8], lrow[8];
        #pragma unroll
        for (int i = 0; i < 8; ++i) { mrow[i] = -1e30f; lrow[i] = 0.0f; }

        for (int kb = 0; kb < 8; ++kb) {
            const int kt0 = kb * 32;
            v8f s0, s1;
            #pragma unroll
            for (int t = 0; t < 2; ++t) {
                const int key = kt0 + t * 16 + l16;
                v16h bfrag = cat8(*(const v8h*)&Ks[key * LDKS + g * 8],
                                  *(const v8h*)&Ks[key * LDKS + 16 + g * 8]);
                v8f c;   // additive mask rides in the accumulator
                #pragma unroll
                for (int i = 0; i < 8; ++i)
                    c[i] = mbase[(q0 + i + g * 8) * 256 + key];
                v8f s = wmma_f16(afrag, bfrag, c);
                if (t == 0) s0 = s; else s1 = s;
            }
            // online softmax over this 32-key block (row = 16 lanes of half-wave)
            #pragma unroll
            for (int i = 0; i < 8; ++i) {
                float tmax = fmaxf(s0[i], s1[i]);
                tmax = fmaxf(tmax, __shfl_xor(tmax, 1, 32));
                tmax = fmaxf(tmax, __shfl_xor(tmax, 2, 32));
                tmax = fmaxf(tmax, __shfl_xor(tmax, 4, 32));
                tmax = fmaxf(tmax, __shfl_xor(tmax, 8, 32));
                const float newm = fmaxf(mrow[i], tmax);
                const float corr = __expf(mrow[i] - newm);
                mrow[i] = newm;
                const float p0 = __expf(s0[i] - newm);
                const float p1 = __expf(s1[i] - newm);
                float rs = p0 + p1;
                rs += __shfl_xor(rs, 1, 32);
                rs += __shfl_xor(rs, 2, 32);
                rs += __shfl_xor(rs, 4, 32);
                rs += __shfl_xor(rs, 8, 32);
                lrow[i] = lrow[i] * corr + rs;
                o0[i] *= corr;  o1[i] *= corr;
                const int prow = i + g * 8;          // C-layout -> scratch
                myPs[prow * LDPS + l16]      = (half_t)p0;
                myPs[prow * LDPS + 16 + l16] = (half_t)p1;
            }
            // P back as A-fragment (same-wave LDS RAW, in-order)
            const half_t* pr = &myPs[l16 * LDPS];
            const v16h pfrag = cat8(*(const v8h*)(pr + g * 8),
                                    *(const v8h*)(pr + 16 + g * 8));
            #pragma unroll
            for (int t = 0; t < 2; ++t) {
                const half_t* vp = &VTs[(t * 16 + l16) * LDVT + kt0 + g * 8];
                v16h vfrag = cat8(*(const v8h*)vp, *(const v8h*)(vp + 16));
                if (t == 0) o0 = wmma_f16(pfrag, vfrag, o0);
                else        o1 = wmma_f16(pfrag, vfrag, o1);
            }
        }
        // finalize rows, store O directly in (b, n, c) layout
        #pragma unroll
        for (int i = 0; i < 8; ++i) {
            const float inv = 1.0f / lrow[i];
            const int n = q0 + i + g * 8;
            const long orow = ((long)b * 256 + n) * 256 + h * 32;
            O[orow + l16]      = (half_t)(o0[i] * inv);
            O[orow + 16 + l16] = (half_t)(o1[i] * inv);
        }
    }
}

// ---------------------------------------------------------------------------
// Kernel 3: output projection  out = O(131072x256) @ Wp(256x256) + bp  (f32)
// A tile staged by TDM (pure f16 copy, 512B rows padded to 528B pitch).
// Block tile 64x64, each wave 16x32.
// ---------------------------------------------------------------------------
__global__ __launch_bounds__(256) void proj_kernel(
    const half_t* __restrict__ O, const float* __restrict__ Wp,
    const float* __restrict__ bp, float* __restrict__ out)
{
    __shared__ half_t As [64 * LDK];
    __shared__ half_t wsT[64 * LDK];

    const int tid  = threadIdx.x;
    const int lane = tid & 31;
    const int w    = tid >> 5;
    const int g    = lane >> 4;
    const int l16  = lane & 15;
    const long rowBase = (long)blockIdx.x * 64;
    const int  colBase = blockIdx.y * 64;

    // TDM: A tile 64 rows x 256 halfs, 512B rows padded to 528B pitch.
    if (w == 0) {
        tdm_load_2d_f16(O + rowBase * 256,
                        (unsigned)(uintptr_t)(void*)&As[0],
                        /*tdim0=*/256, /*tdim1=*/64, /*stride0=*/256,
                        /*tile0=*/256, /*tile1=*/64,
                        /*pad_interval(128 dw)=*/6u, /*pad_amount(4 dw)=*/3u);
        __builtin_amdgcn_s_wait_tensorcnt(0);
    }
    #pragma unroll
    for (int i = 0; i < 64; ++i) {
        int idx = tid + i * 256;
        int k = idx >> 6, col = idx & 63;
        wsT[col * LDK + k] = (half_t)Wp[(long)k * 256 + colBase + col];
    }
    __syncthreads();

    const int wm = w >> 1, wn = w & 1;
    const int mBase = wm * 16, nBase = wn * 32;
    const int gcol0 = colBase + nBase + l16;
    const float bias0 = bp[gcol0];
    const float bias1 = bp[gcol0 + 16];
    v8f acc0 = { bias0,bias0,bias0,bias0,bias0,bias0,bias0,bias0 };
    v8f acc1 = { bias1,bias1,bias1,bias1,bias1,bias1,bias1,bias1 };

    const half_t* arow  = &As[(mBase + l16) * LDK];
    const half_t* brow0 = &wsT[(nBase + l16) * LDK];
    const half_t* brow1 = &wsT[(nBase + 16 + l16) * LDK];
    #pragma unroll
    for (int kk = 0; kk < 8; ++kk) {
        const int k0 = kk * 32;
        v16h a  = cat8(*(const v8h*)(arow + k0 + g * 8),
                       *(const v8h*)(arow + k0 + 16 + g * 8));
        v16h b0 = cat8(*(const v8h*)(brow0 + k0 + g * 8),
                       *(const v8h*)(brow0 + k0 + 16 + g * 8));
        v16h b1 = cat8(*(const v8h*)(brow1 + k0 + g * 8),
                       *(const v8h*)(brow1 + k0 + 16 + g * 8));
        acc0 = wmma_f16(a, b0, acc0);
        acc1 = wmma_f16(a, b1, acc1);
    }
    #pragma unroll
    for (int i = 0; i < 8; ++i) {
        const long gr = rowBase + mBase + g * 8 + i;
        out[gr * 256 + gcol0]      = acc0[i];
        out[gr * 256 + gcol0 + 16] = acc1[i];
    }
}

// ---------------------------------------------------------------------------
extern "C" void kernel_launch(void* const* d_in, const int* in_sizes, int n_in,
                              void* d_out, int out_size, void* d_ws, size_t ws_size,
                              hipStream_t stream) {
    (void)in_sizes; (void)n_in; (void)out_size; (void)ws_size;
    const float* x    = (const float*)d_in[0];
    const float* mask = (const float*)d_in[1];
    const float* Wq   = (const float*)d_in[2];
    const float* bq   = (const float*)d_in[3];
    const float* Wkv  = (const float*)d_in[4];
    const float* bkv  = (const float*)d_in[5];
    const float* Wp   = (const float*)d_in[6];
    const float* bp   = (const float*)d_in[7];
    float* out = (float*)d_out;

    half_t* ws = (half_t*)d_ws;
    const long QE = 33554432L;            // 512*8*256*32 elements
    half_t* Q = ws;
    half_t* K = ws + QE;
    half_t* V = ws + 2 * QE;
    half_t* O = ws + 3 * QE;              // (b, n, c) f16

    qkv_kernel <<<dim3(2048, 12), 256, 0, stream>>>(x, Wq, bq, Wkv, bkv, Q, K, V);
    attn_kernel<<<dim3(4096, 1),  256, 0, stream>>>(Q, K, V, mask, O);
    proj_kernel<<<dim3(2048, 4),  256, 0, stream>>>(O, Wp, bp, out);
}
